// MLA_2903397892755
// MI455X (gfx1250) — matrix operand order repaired and checked by reading, not compiled, not run on hardware
//
#include <hip/hip_runtime.h>
#include <hip/hip_bf16.h>

typedef __attribute__((ext_vector_type(16))) __bf16 v16bf;
typedef __attribute__((ext_vector_type(8)))  __bf16 v8bf;
typedef __attribute__((ext_vector_type(4)))  __bf16 v4bf;
typedef __attribute__((ext_vector_type(8)))  float  v8f;
typedef __attribute__((ext_vector_type(4)))  float  v4f;

// Problem constants (match reference)
constexpr int Bc    = 2;
constexpr int Sc    = 1024;
constexpr int DIMc  = 2048;
constexpr int Hc    = 16;
constexpr int NOPEc = 128;
constexpr int ROPEc = 64;
constexpr int VDc   = 128;
constexpr int QKDc  = NOPEc + ROPEc; // 192

// ---------------------------------------------------------------------------
// WMMA + fragment helpers (CDNA5 bf16 16x16x32, f32 accumulate)
// ---------------------------------------------------------------------------
__device__ __forceinline__ v8f wmma_bf16(v16bf a, v16bf b, v8f c) {
  return __builtin_amdgcn_wmma_f32_16x16x32_bf16(
      /*neg_a=*/false, a, /*neg_b=*/false, b,
      /*c_mod=*/(short)0, c, /*reuse_a=*/false, /*reuse_b=*/false);
}

__device__ __forceinline__ v16bf cat16(v8bf a, v8bf b) {
  return __builtin_shufflevector(a, b, 0, 1, 2, 3, 4, 5, 6, 7, 8, 9, 10, 11,
                                 12, 13, 14, 15);
}

// A-fragment 16x32 from row-major bf16 (row stride ld, 16B-aligned rows).
// Lane m=lane&15 owns row m; elements k = hi*8+[0..7] and 16+hi*8+[0..7].
__device__ __forceinline__ v16bf load_a_ct(const __bf16* p, int ld) {
  const int lane = threadIdx.x & 31;
  const int m = lane & 15, hi = (lane >> 4) & 1;
  const __bf16* q = p + (size_t)m * ld + hi * 8;
  return cat16(*(const v8bf*)q, *(const v8bf*)(q + 16));
}

// B-fragment 32x16 from k-contiguous storage: B[k][n] = p[n*ld + k].
// Lane n=lane&15 owns column n; elements k = hi*8+[0..7] and 16+hi*8+[0..7].
__device__ __forceinline__ v16bf load_b_ct(const __bf16* p, int ld) {
  const int lane = threadIdx.x & 31;
  const int n = lane & 15, hi = (lane >> 4) & 1;
  const __bf16* q = p + (size_t)n * ld + hi * 8;
  return cat16(*(const v8bf*)q, *(const v8bf*)(q + 16));
}

// ---------------------------------------------------------------------------
// Tiled GEMM: C(MxN) = A(MxK,f32) @ B(KxN,f32), bf16 compute, f32 accumulate.
// Block tile 128x128, BK=32, 256 threads (8 waves), wave tile 32x64 (2x4
// WMMA accumulators). Double-buffered LDS + next-tile prefetch.
// MODE 0: bf16 out row-major; MODE 1: f32 out + bias;
// MODE 2: bf16 out transposed to [b][h][d][s] (for V).
// ---------------------------------------------------------------------------
#define GBM 128
#define GBN 128
#define GBK 32
#define LDA (GBK + 8)  // 40 elems = 80B rows (16B aligned)
#define LDB (GBK + 8)  // transposed B tile: [n][k], 40-elem rows

// Stage one (A,B) k-tile into LDS (f32 -> bf16, B transposed to k-contiguous).
__device__ __forceinline__ void stage_tiles(const float* __restrict__ A,
                                            const float* __restrict__ Bw,
                                            int N, int K, int row0, int col0,
                                            int k0, int t,
                                            __bf16 (*sA)[LDA],
                                            __bf16 (*sBt)[LDB]) {
  // A: 128x32 f32 -> bf16, float4 loads + b64 LDS stores
#pragma unroll
  for (int i = 0; i < 4; ++i) {
    const int idx4 = i * 256 + t;            // 1024 float4s
    const int r = idx4 >> 3, c4 = idx4 & 7;  // 8 float4 per row
    const v4f a = *(const v4f*)(A + (size_t)(row0 + r) * K + k0 + c4 * 4);
    v4bf b;
    b[0] = (__bf16)a[0]; b[1] = (__bf16)a[1];
    b[2] = (__bf16)a[2]; b[3] = (__bf16)a[3];
    *(v4bf*)&sA[r][c4 * 4] = b;
  }
  // B: 32x128, coalesced float4 global reads, transposed b16 LDS stores
#pragma unroll
  for (int i = 0; i < 4; ++i) {
    const int idx4 = i * 256 + t;             // 1024 float4s
    const int r = idx4 >> 5, c4 = idx4 & 31;  // 32 float4 per 128-col row
    const v4f a = *(const v4f*)(Bw + (size_t)(k0 + r) * N + col0 + c4 * 4);
#pragma unroll
    for (int j = 0; j < 4; ++j) sBt[c4 * 4 + j][r] = (__bf16)a[j];
  }
}

template <int MODE>
__global__ __launch_bounds__(256) void mla_gemm(
    const float* __restrict__ A, const float* __restrict__ Bw,
    void* __restrict__ Cout, const float* __restrict__ bias, int M, int N,
    int K) {
  __shared__ alignas(16) __bf16 sA[2][GBM][LDA];
  __shared__ alignas(16) __bf16 sBt[2][GBN][LDB];  // transposed: [n][k]

  const int t = threadIdx.x;
  const int lane = t & 31;
  const int w = t >> 5;
  const int wm = w & 3;   // 0..3 -> 32-row band
  const int wn = w >> 2;  // 0..1 -> 64-col band
  const int row0 = blockIdx.y * GBM;
  const int col0 = blockIdx.x * GBN;
  const int nlo = lane & 15, hi = (lane >> 4) & 1;

  v8f acc[2][4] = {};

  stage_tiles(A, Bw, N, K, row0, col0, 0, t, sA[0], sBt[0]);

  for (int k0 = 0; k0 < K; k0 += GBK) {
    const int cur = (k0 / GBK) & 1;
    __syncthreads();
    if (k0 + GBK < K) {
      stage_tiles(A, Bw, N, K, row0, col0, k0 + GBK, t, sA[cur ^ 1],
                  sBt[cur ^ 1]);
      if (k0 + 2 * GBK < K) {
        // speculative prefetch of the tile after next into GL2
        __builtin_prefetch(A + (size_t)(row0 + (t >> 1)) * K + k0 + 2 * GBK, 0,
                           1);
        __builtin_prefetch(Bw + (size_t)(k0 + 2 * GBK + (t >> 3)) * N + col0 +
                               (t & 7) * 16,
                           0, 1);
      }
    }

    v16bf af[2], bf[4];
#pragma unroll
    for (int i = 0; i < 2; ++i)
      af[i] = load_a_ct(&sA[cur][wm * 32 + i * 16][0], LDA);
#pragma unroll
    for (int j = 0; j < 4; ++j)
      bf[j] = load_b_ct(&sBt[cur][wn * 64 + j * 16][0], LDB);
#pragma unroll
    for (int i = 0; i < 2; ++i)
#pragma unroll
      for (int j = 0; j < 4; ++j) acc[i][j] = wmma_bf16(af[i], bf[j], acc[i][j]);
  }

  // Epilogue: VGPR r -> row r (lanes 0-15) / r+8 (lanes 16-31), col = lane&15
#pragma unroll
  for (int i = 0; i < 2; ++i) {
#pragma unroll
    for (int j = 0; j < 4; ++j) {
#pragma unroll
      for (int r = 0; r < 8; ++r) {
        const int row = row0 + wm * 32 + i * 16 + r + 8 * hi;
        const int col = col0 + wn * 64 + j * 16 + nlo;
        const float v = acc[i][j][r];
        if constexpr (MODE == 0) {
          ((__bf16*)Cout)[(size_t)row * N + col] = (__bf16)v;
        } else if constexpr (MODE == 1) {
          ((float*)Cout)[(size_t)row * N + col] = v + bias[col];
        } else {
          // V: row = b*Sc + s, col = h*VDc + d  ->  [b][h][d][s]
          const int b = row >> 10, s = row & (Sc - 1);
          const int h = col >> 7, d = col & (VDc - 1);
          ((__bf16*)Cout)[(((size_t)(b * Hc + h) * VDc + d) * Sc) + s] =
              (__bf16)v;
        }
      }
    }
  }
}

// ---------------------------------------------------------------------------
// RoPE on the rope slice of q and k (bf16 in place, f32 math).
// ---------------------------------------------------------------------------
__global__ __launch_bounds__(256) void mla_rope(__bf16* __restrict__ q,
                                                __bf16* __restrict__ k,
                                                const int* __restrict__ sp_ptr) {
  const int idx = blockIdx.x * blockDim.x + threadIdx.x;
  constexpr int HALF = ROPEc / 2; // 32
  const int i = idx % HALF;
  int rest = idx / HALF;
  const int h = rest % Hc;
  rest /= Hc;
  const int s = rest % Sc;
  const int b = rest / Sc;
  if (b >= Bc) return;

  const int sp = *sp_ptr;
  const float theta = __powf(10000.0f, -(float)(2 * i) / (float)ROPEc);
  const float ang = (float)(s + sp) * theta;
  const float c = __cosf(ang), sn = __sinf(ang);

  const size_t base = (((size_t)(b * Sc + s) * Hc + h) * QKDc) + NOPEc + 2 * i;

  float q0 = (float)q[base], q1 = (float)q[base + 1];
  q[base]     = (__bf16)(q0 * c - q1 * sn);
  q[base + 1] = (__bf16)(q1 * c + q0 * sn);

  float k0 = (float)k[base], k1 = (float)k[base + 1];
  k[base]     = (__bf16)(k0 * c - k1 * sn);
  k[base + 1] = (__bf16)(k1 * c + k0 * sn);
}

// ---------------------------------------------------------------------------
// Causal flash attention: one wave per (b, h, 16-query tile).
// q/k row-major [b][s][h][d]; v transposed [b][h][d][s] so every fragment
// load is a pair of 16B vector loads.
// ---------------------------------------------------------------------------
__global__ __launch_bounds__(32) void mla_attn(const __bf16* __restrict__ qb,
                                               const __bf16* __restrict__ kb,
                                               const __bf16* __restrict__ vt,
                                               float* __restrict__ cv) {
  __shared__ alignas(16) __bf16 lds_p[16 * 32];  // P staging (C -> A layout)

  const int lane = threadIdx.x & 31;
  const int nlo = lane & 15, hi = (lane >> 4) & 1;
  const int qt = blockIdx.x;  // 0..S/16-1
  const int h  = blockIdx.y;  // 0..H-1
  const int b  = blockIdx.z;  // 0..B-1
  const int ldq = Hc * QKDc;  // token stride in q/k

  // Preload Q fragments (A layout; k-contiguous in global)
  v16bf qf[6];
  {
    const __bf16* qp =
        qb + (((size_t)(b * Sc) + qt * 16 + nlo) * Hc + h) * QKDc + hi * 8;
#pragma unroll
    for (int c = 0; c < 6; ++c)
      qf[c] = cat16(*(const v8bf*)(qp + c * 32),
                    *(const v8bf*)(qp + c * 32 + 16));
  }

  v8f oacc[8] = {};        // 16 x 128 output accumulator
  float mrow[8], lrow[8];  // per-row running max / sum (rows r+8*hi)
#pragma unroll
  for (int r = 0; r < 8; ++r) { mrow[r] = -1e30f; lrow[r] = 0.0f; }

  const float scale = rsqrtf((float)QKDc);
  const int qmax = qt * 16 + 15;
  const __bf16* vbase = vt + ((size_t)(b * Hc) + h) * VDc * Sc;

  for (int j0 = 0; j0 <= qmax; j0 += 32) {
    // ---- scores: two 16x16 tiles (keys j0..j0+15, j0+16..j0+31) ----
    v8f s0 = {}, s1 = {};
    const __bf16* kb0 = kb + (((size_t)(b * Sc) + j0) * Hc + h) * QKDc;
#pragma unroll
    for (int c = 0; c < 6; ++c) {
      s0 = wmma_bf16(qf[c], load_b_ct(kb0 + c * 32, ldq), s0);
      s1 = wmma_bf16(qf[c], load_b_ct(kb0 + 16 * ldq + c * 32, ldq), s1);
    }

    // ---- scale + causal mask + online softmax ----
    float p0[8], p1[8], alpha[8];
#pragma unroll
    for (int r = 0; r < 8; ++r) {
      const int qrow = qt * 16 + r + 8 * hi;
      const int c0 = j0 + nlo, c1 = j0 + 16 + nlo;
      float v0 = s0[r] * scale + ((c0 > qrow) ? -1e9f : 0.0f);
      float v1 = s1[r] * scale + ((c1 > qrow) ? -1e9f : 0.0f);
      float mx = fmaxf(v0, v1);
#pragma unroll
      for (int off = 8; off >= 1; off >>= 1)
        mx = fmaxf(mx, __shfl_xor(mx, off));
      const float mnew = fmaxf(mrow[r], mx);
      alpha[r] = __expf(mrow[r] - mnew);
      p0[r] = __expf(v0 - mnew);
      p1[r] = __expf(v1 - mnew);
      float rs = p0[r] + p1[r];
#pragma unroll
      for (int off = 8; off >= 1; off >>= 1) rs += __shfl_xor(rs, off);
      lrow[r] = lrow[r] * alpha[r] + rs;
      mrow[r] = mnew;
    }
#pragma unroll
    for (int n = 0; n < 8; ++n)
#pragma unroll
      for (int r = 0; r < 8; ++r) oacc[n][r] *= alpha[r];

    // ---- stage P (16x32) through LDS, reload as A-fragment ----
    __syncthreads();
#pragma unroll
    for (int r = 0; r < 8; ++r) {
      lds_p[(r + 8 * hi) * 32 + nlo]      = (__bf16)p0[r];
      lds_p[(r + 8 * hi) * 32 + 16 + nlo] = (__bf16)p1[r];
    }
    __syncthreads();
    const v16bf pf = load_a_ct(lds_p, 32);

    // ---- P @ V : 8 column slabs of VD=128 (V is d-major, key-contiguous) ----
#pragma unroll
    for (int n = 0; n < 8; ++n) {
      const v16bf vf = load_b_ct(vbase + (size_t)(n * 16) * Sc + j0, Sc);
      oacc[n] = wmma_bf16(pf, vf, oacc[n]);
    }
    __syncthreads();
  }

  // ---- normalize and store c_v (f32, row-major [b][s][h][d]) ----
  float inv[8];
#pragma unroll
  for (int r = 0; r < 8; ++r) inv[r] = 1.0f / lrow[r];
#pragma unroll
  for (int n = 0; n < 8; ++n) {
#pragma unroll
    for (int r = 0; r < 8; ++r) {
      const int row = qt * 16 + r + 8 * hi;
      cv[(((size_t)(b * Sc) + row) * Hc + h) * VDc + n * 16 + nlo] =
          oacc[n][r] * inv[r];
    }
  }
}

// ---------------------------------------------------------------------------
// Host-side launch
// ---------------------------------------------------------------------------
extern "C" void kernel_launch(void* const* d_in, const int* in_sizes, int n_in,
                              void* d_out, int out_size, void* d_ws,
                              size_t ws_size, hipStream_t stream) {
  (void)in_sizes; (void)n_in; (void)out_size; (void)ws_size;

  const float* x  = (const float*)d_in[0];
  const float* Wq = (const float*)d_in[1];
  const float* Wk = (const float*)d_in[2];
  const float* Wv = (const float*)d_in[3];
  const float* Wo = (const float*)d_in[4];
  const float* bo = (const float*)d_in[5];
  // d_in[6] = mask (implemented analytically), d_in[7] = start_pos
  const int* start_pos = (const int*)d_in[7];
  float* out = (float*)d_out;

  // Workspace carve-out
  __bf16* qbf = (__bf16*)d_ws;
  __bf16* kbf = qbf + (size_t)Bc * Sc * Hc * QKDc;
  __bf16* vbt = kbf + (size_t)Bc * Sc * Hc * QKDc;  // [b][h][d][s]
  float*  cv  = (float*)(vbt + (size_t)Bc * Sc * Hc * VDc);

  const int M  = Bc * Sc;     // 2048
  const int NQ = Hc * QKDc;   // 3072
  const int NV = Hc * VDc;    // 2048

  dim3 blk(256);
  // Q/K/V projections (f32 in -> bf16 out)
  mla_gemm<0><<<dim3(NQ / GBN, M / GBM), blk, 0, stream>>>(
      x, Wq, qbf, nullptr, M, NQ, DIMc);
  mla_gemm<0><<<dim3(NQ / GBN, M / GBM), blk, 0, stream>>>(
      x, Wk, kbf, nullptr, M, NQ, DIMc);
  mla_gemm<2><<<dim3(NV / GBN, M / GBM), blk, 0, stream>>>(
      x, Wv, vbt, nullptr, M, NV, DIMc);

  // RoPE on rope slices of q, k
  {
    const int total = Bc * Sc * Hc * (ROPEc / 2);
    mla_rope<<<dim3((total + 255) / 256), blk, 0, stream>>>(qbf, kbf,
                                                            start_pos);
  }

  // Causal flash attention -> c_v (f32)
  mla_attn<<<dim3(Sc / 16, Hc, Bc), dim3(32), 0, stream>>>(qbf, kbf, vbt, cv);

  // Output projection with bias (f32 out)
  mla_gemm<1><<<dim3(DIMc / GBN, M / GBM), blk, 0, stream>>>(
      cv, Wo, out, bo, M, DIMc, NV);
}
